// LSTM_AE_90426241450325
// MI455X (gfx1250) — matrix operand order repaired and checked by reading, not compiled, not run on hardware
//
#include <hip/hip_runtime.h>
#include <stdint.h>

// ---------------- problem dims ----------------
#define B_   128
#define T_   1024
#define F_   128
#define H_   512
#define G4H_ 2048

#define NBLK 32     // persistent grid: each block owns 16 h-columns
#define NTHR 256    // 8 wave32 per block: wave w owns row-tile w (16 rows of B_=128)

// padded LDS row strides (elements) to spread fragment loads across banks
#define SIN_STRIDE  136   // 128 + 8  (TDM pad: interval code 5 = 64 DW, amount code 3 = 4 DW)
#define SW_STRIDE   520   // 512 + 8  (TDM pad: interval code 7 = 256 DW, amount code 3 = 4 DW)
#define SMEM_ELEMS  (B_ * SIN_STRIDE + 64 * SW_STRIDE + 16 * SW_STRIDE)  // 59008 -> 118016 B

// ---------------- vector types ----------------
typedef __attribute__((ext_vector_type(16))) __bf16         bf16x16;
typedef __attribute__((ext_vector_type(8)))  float          f32x8;
typedef __attribute__((ext_vector_type(8)))  unsigned short u16x8;
typedef __attribute__((ext_vector_type(4)))  unsigned int   u32x4;
typedef __attribute__((ext_vector_type(8)))  int            i32x8;
typedef __attribute__((ext_vector_type(4)))  int            i32x4;

union FragU { u16x8 h[2]; bf16x16 v; };

#define WMMA_BF16(a, b, c) \
  __builtin_amdgcn_wmma_f32_16x16x32_bf16(false, (a), false, (b), (short)0, (c), false, false)

// ---------------- helpers ----------------
__device__ __forceinline__ unsigned short f32_to_bf16_rne(float f) {
  unsigned u = __float_as_uint(f);
  u += 0x7FFFu + ((u >> 16) & 1u);
  return (unsigned short)(u >> 16);
}
__device__ __forceinline__ float sigm(float x) { return 1.0f / (1.0f + __expf(-x)); }

// A fragment (16x32 bf16, M x K): lane holds row M=lane%16; VGPR0-3 = K[g*8..g*8+7],
// VGPR4-7 = K[16+g*8..+7], g = lane/16.  Row-major source, stride in elements.
__device__ __forceinline__ bf16x16 frag_a_global(const unsigned short* base, int row,
                                                 int stride, int k0, int g) {
  const unsigned short* p = base + (size_t)row * stride + k0 + g * 8;
  FragU f;
  f.h[0] = *(const u16x8*)(p);
  f.h[1] = *(const u16x8*)(p + 16);
  return f.v;
}
// B fragment (32x16 bf16, K x N): lane holds column N=lane%16, K=[g*8..g*8+7],[16+g*8..+7]
// (table order).  Weight row-major [N,K] -> column-major from WMMA's view (global source).
__device__ __forceinline__ bf16x16 frag_b_global(const unsigned short* base, int n,
                                                 int stride, int k0, int g) {
  const unsigned short* p = base + (size_t)n * stride + k0 + g * 8;
  FragU f;
  f.h[0] = *(const u16x8*)(p);
  f.h[1] = *(const u16x8*)(p + 16);
  return f.v;
}
// B fragment from an LDS-resident padded weight row (rowp points at row start).
__device__ __forceinline__ bf16x16 frag_b_lds(const unsigned short* rowp, int k0, int g) {
  const unsigned short* p = rowp + k0 + g * 8;
  FragU f;
  f.h[0] = *(const u16x8*)(p);
  f.h[1] = *(const u16x8*)(p + 16);
  return f.v;
}

// ---------------- TDM staging (global -> LDS) ----------------
#if defined(__has_builtin)
#  if __has_builtin(__builtin_amdgcn_tensor_load_to_lds)
#    define HAVE_TDM 1
#  endif
#endif

#ifdef HAVE_TDM
// 2D tile load: rows x cols bf16 elements, row stride in elements, optional LDS padding.
// D# per CDNA5 ISA §8: group0 = {count|flags, lds_addr, global_addr, type=2},
// group1 = {data_size | pad ctl, tensor dims/tile dims/strides}.
__device__ __forceinline__ void tdm_load_2d(unsigned lds_off, const unsigned short* gptr,
                                            unsigned cols, unsigned rows,
                                            unsigned long long row_stride_elems,
                                            unsigned pad_interval_code,
                                            unsigned pad_amount_code) {
  unsigned long long ga = (unsigned long long)(uintptr_t)gptr;
  u32x4 g0;
  g0.x = 1u;                                               // count=1, user mode
  g0.y = lds_off;                                          // lds_addr (bytes)
  g0.z = (unsigned)ga;                                     // global_addr[31:0]
  g0.w = (unsigned)((ga >> 32) & 0x1FFFFFFu) | (2u << 30); // addr[56:32] | type=2
  i32x8 g1;
  g1[0] = (int)((1u << 16)                                 // data_size=1 -> 2 bytes
              | (1u << 20)                                 // pad_enable
              | (pad_interval_code << 22)
              | (pad_amount_code << 25));
  g1[1] = (int)((cols & 0xFFFFu) << 16);                   // tensor_dim0[15:0]
  g1[2] = (int)(((cols >> 16) & 0xFFFFu) | ((rows & 0xFFFFu) << 16)); // dim0 hi | dim1 lo
  g1[3] = (int)(((rows >> 16) & 0xFFFFu) | ((cols & 0xFFFFu) << 16)); // dim1 hi | tile_dim0
  g1[4] = (int)(rows & 0xFFFFu);                           // tile_dim1 | tile_dim2=0
  g1[5] = (int)(unsigned)(row_stride_elems & 0xFFFFFFFFull);          // dim0_stride lo
  g1[6] = (int)(unsigned)((row_stride_elems >> 32) & 0xFFFFull);      // stride hi | s1 lo=0
  g1[7] = 0;
  i32x4 z4 = {0, 0, 0, 0};
#if __clang_major__ >= 23
  i32x8 z8 = {0, 0, 0, 0, 0, 0, 0, 0};
  __builtin_amdgcn_tensor_load_to_lds(g0, g1, z4, z4, z8, 0);
#else
  __builtin_amdgcn_tensor_load_to_lds(g0, g1, z4, z4, 0);
#endif
}
#endif

// Fallback cooperative copy into padded LDS rows (also host-pass body).
__device__ __forceinline__ void copy_padded(unsigned short* dst, int dstStride,
                                            const unsigned short* src, long long srcStride,
                                            int rows, int cols, int tid) {
  const int cpr = cols / 8;
  for (int i = tid; i < rows * cpr; i += NTHR) {
    int r = i / cpr, c = (i % cpr) * 8;
    *(u16x8*)&dst[r * dstStride + c] = *(const u16x8*)&src[(long long)r * srcStride + c];
  }
}

// Stage one [B_ x F_] input tile (row stride in elements) into padded sIn.
__device__ __forceinline__ void stage_input(unsigned short* sIn, const unsigned short* gptr,
                                            unsigned long long row_stride_elems, int tid) {
#ifdef HAVE_TDM
  if ((tid >> 5) == 0) {   // wave-uniform: wave 0 issues the DMA (EXEC ignored by TDM)
    tdm_load_2d((unsigned)(uintptr_t)sIn, gptr, F_, B_, row_stride_elems, 5u, 3u);
    __builtin_amdgcn_s_wait_tensorcnt(0);
  }
  __syncthreads();
#else
  copy_padded(sIn, SIN_STRIDE, gptr, (long long)row_stride_elems, B_, F_, tid);
  __syncthreads();
#endif
}

// Stage this block's Whh slice (4 gates x 16 cols x 512 K) and, for blocks 0..7,
// the Wout slice (16 rows x 512 K) into padded LDS.  Done once per LSTM direction.
__device__ __forceinline__ void stage_weights(unsigned short* sWhh, unsigned short* sWout,
                                              const unsigned short* Whh,
                                              const unsigned short* Wout,  // may be nullptr
                                              int col0, int wg, int tid) {
#ifdef HAVE_TDM
  if ((tid >> 5) == 0) {
    for (int gi = 0; gi < 4; ++gi) {
      tdm_load_2d((unsigned)(uintptr_t)(sWhh + gi * 16 * SW_STRIDE),
                  Whh + (size_t)(gi * H_ + col0) * H_, H_, 16, H_, 7u, 3u);
    }
    if (Wout && wg < F_ / 16) {
      tdm_load_2d((unsigned)(uintptr_t)sWout,
                  Wout + (size_t)(wg * 16) * H_, H_, 16, H_, 7u, 3u);
    }
    __builtin_amdgcn_s_wait_tensorcnt(0);
  }
  __syncthreads();
#else
  for (int gi = 0; gi < 4; ++gi)
    copy_padded(sWhh + gi * 16 * SW_STRIDE, SW_STRIDE,
                Whh + (size_t)(gi * H_ + col0) * H_, H_, 16, H_, tid);
  if (Wout && wg < F_ / 16)
    copy_padded(sWout, SW_STRIDE, Wout + (size_t)(wg * 16) * H_, H_, 16, H_, tid);
  __syncthreads();
#endif
}

// ---------------- grid-wide barrier (persistent kernel, 32 resident blocks) -------------
__device__ __forceinline__ void grid_barrier(unsigned* bar) {
  __threadfence();
  __syncthreads();
  if (threadIdx.x == 0) {
    unsigned old = __hip_atomic_fetch_add(bar, 1u, __ATOMIC_ACQ_REL, __HIP_MEMORY_SCOPE_AGENT);
    unsigned target = (old / NBLK + 1u) * NBLK;   // monotone generation counter
    while (__hip_atomic_load(bar, __ATOMIC_ACQUIRE, __HIP_MEMORY_SCOPE_AGENT) < target) {
      __builtin_amdgcn_s_sleep(8);
    }
  }
  __syncthreads();
}

// ---------------- one LSTM gate step for this block's 16 h-columns ----------------
// gates[128, 2048] = sIn[128,128] x Wih^T + hcur[128,512] x Whh^T + bias
// Wih B-fragments stay in VGPRs (compiler hoists them across the time loop);
// Whh B-fragments come from the LDS-resident slice; A-hidden comes from L2-resident h.
__device__ __forceinline__ void gate_phase(
    const unsigned short* sIn,          // LDS: [B_ x SIN_STRIDE] input tile (bf16)
    const unsigned short* sWhh,         // LDS: [64 x SW_STRIDE] Whh slice (bf16)
    const unsigned short* hcur,         // [B_ x H_] bf16 h(t-1)
    const unsigned short* Wih,          // [4H x F] bf16 (global; hoisted to VGPRs)
    const float* bias,                  // [4H] combined bih+bhh
    float* c_buf,                       // [B_ x H_] fp32 cell state (block-exclusive cols)
    unsigned short* hnext,              // [B_ x H_] bf16 h(t) output buffer
    int col0, int wv, int lane)
{
  const int nL  = lane & 15;
  const int g   = lane >> 4;
  const int row = wv * 16 + nL;

  f32x8 acc0, acc1, acc2, acc3;
  {
    float b0 = bias[0 * H_ + col0 + nL];
    float b1 = bias[1 * H_ + col0 + nL];
    float b2 = bias[2 * H_ + col0 + nL];
    float b3 = bias[3 * H_ + col0 + nL];
    for (int r = 0; r < 8; ++r) { acc0[r] = b0; acc1[r] = b1; acc2[r] = b2; acc3[r] = b3; }
  }

  // input contribution (K = F_ = 128), A from LDS, B hoisted from global Wih
  for (int k0 = 0; k0 < F_; k0 += 32) {
    FragU fa;
    fa.h[0] = *(const u16x8*)&sIn[row * SIN_STRIDE + k0 + g * 8];
    fa.h[1] = *(const u16x8*)&sIn[row * SIN_STRIDE + k0 + g * 8 + 16];
    bf16x16 a = fa.v;
    acc0 = WMMA_BF16(a, frag_b_global(Wih, 0 * H_ + col0 + nL, F_, k0, g), acc0);
    acc1 = WMMA_BF16(a, frag_b_global(Wih, 1 * H_ + col0 + nL, F_, k0, g), acc1);
    acc2 = WMMA_BF16(a, frag_b_global(Wih, 2 * H_ + col0 + nL, F_, k0, g), acc2);
    acc3 = WMMA_BF16(a, frag_b_global(Wih, 3 * H_ + col0 + nL, F_, k0, g), acc3);
  }
  // hidden contribution (K = H_ = 512): A from L2-resident h, B from LDS slice
  const unsigned short* r0 = sWhh + (0 * 16 + nL) * SW_STRIDE;
  const unsigned short* r1 = sWhh + (1 * 16 + nL) * SW_STRIDE;
  const unsigned short* r2 = sWhh + (2 * 16 + nL) * SW_STRIDE;
  const unsigned short* r3 = sWhh + (3 * 16 + nL) * SW_STRIDE;
  for (int k0 = 0; k0 < H_; k0 += 32) {
    bf16x16 a = frag_a_global(hcur, row, H_, k0, g);
    acc0 = WMMA_BF16(a, frag_b_lds(r0, k0, g), acc0);
    acc1 = WMMA_BF16(a, frag_b_lds(r1, k0, g), acc1);
    acc2 = WMMA_BF16(a, frag_b_lds(r2, k0, g), acc2);
    acc3 = WMMA_BF16(a, frag_b_lds(r3, k0, g), acc3);
  }

  // cell update: C-layout row m = wv*16 + 8*g + r, col n = col0 + lane%16
  const int hc = col0 + nL;
  for (int r = 0; r < 8; ++r) {
    int b   = wv * 16 + g * 8 + r;
    int idx = b * H_ + hc;
    float iv = sigm(acc0[r]);
    float fv = sigm(acc1[r]);
    float gv = tanhf(acc2[r]);
    float ov = sigm(acc3[r]);
    float c  = fv * c_buf[idx] + iv * gv;
    c_buf[idx] = c;
    hnext[idx] = f32_to_bf16_rne(ov * tanhf(c));
  }
}

// ---------------- persistent LSTM-AE kernel ----------------
__global__ __launch_bounds__(NTHR, 1) void lstm_ae_persistent(
    const unsigned short* __restrict__ x_bf,
    const unsigned short* __restrict__ Wih_e, const unsigned short* __restrict__ Whh_e,
    const unsigned short* __restrict__ Wih_d, const unsigned short* __restrict__ Whh_d,
    const unsigned short* __restrict__ Wout,
    const float* __restrict__ bias_e, const float* __restrict__ bias_d,
    const float* __restrict__ bout,
    unsigned short* __restrict__ h_bf,   // 2 * B_ * H_ (double buffered)
    float* __restrict__ c_buf,           // B_ * H_
    unsigned short* __restrict__ y_buf,  // B_ * F_
    float* __restrict__ out,             // B_ * T_ * F_
    unsigned* __restrict__ bar)
{
  extern __shared__ unsigned short smem[];
  unsigned short* sIn   = smem;                          // B_  x SIN_STRIDE
  unsigned short* sWhh  = smem + B_ * SIN_STRIDE;        // 64  x SW_STRIDE
  unsigned short* sWout = sWhh + 64 * SW_STRIDE;         // 16  x SW_STRIDE

  const int tid  = threadIdx.x;
  const int lane = tid & 31;
  const int wv   = tid >> 5;
  const int nL   = lane & 15;
  const int g    = lane >> 4;
  const int wg   = blockIdx.x;
  const int col0 = wg * (H_ / NBLK);        // 16 h-columns per block
  int cur = 0;

  // stage encoder recurrent weights into LDS once
  stage_weights(sWhh, sWout, Whh_e, nullptr, col0, wg, tid);

  // ------------- encoder: 1024 steps -------------
  for (int t = 0; t < T_; ++t) {
    stage_input(sIn, x_bf + (size_t)t * F_, (unsigned long long)T_ * F_, tid);
    if (t + 1 < T_ && tid < B_)   // prefetch next step's input rows into L2/L0
      __builtin_prefetch(&x_bf[(size_t)tid * (T_ * F_) + (size_t)(t + 1) * F_], 0, 1);
    gate_phase(sIn, sWhh, h_bf + cur * (B_ * H_), Wih_e, bias_e,
               c_buf, h_bf + (cur ^ 1) * (B_ * H_), col0, wv, lane);
    cur ^= 1;
    grid_barrier(bar);
  }

  // swap in decoder recurrent weights + output projection
  stage_weights(sWhh, sWout, Whh_d, Wout, col0, wg, tid);

  // ------------- decoder: 1024 steps -------------
  for (int i = 0; i < T_; ++i) {
    const unsigned short* hcur = h_bf + cur * (B_ * H_);

    // y = h @ Wout^T + bout   (blocks 0..7 each own 16 output columns)
    if (wg < F_ / 16) {
      const int f0 = wg * 16;
      f32x8 acc;
      { float bv = bout[f0 + nL]; for (int r = 0; r < 8; ++r) acc[r] = bv; }
      const unsigned short* ro = sWout + nL * SW_STRIDE;
      for (int k0 = 0; k0 < H_; k0 += 32) {
        bf16x16 a = frag_a_global(hcur, wv * 16 + nL, H_, k0, g);
        acc = WMMA_BF16(a, frag_b_lds(ro, k0, g), acc);
      }
      const int fc = f0 + nL;
      for (int r = 0; r < 8; ++r) {
        int b = wv * 16 + g * 8 + r;
        float yv = acc[r];
        out[((size_t)b * T_ + (T_ - 1 - i)) * F_ + fc] = yv;  // flipped emit order
        y_buf[b * F_ + fc] = f32_to_bf16_rne(yv);
      }
    }
    grid_barrier(bar);

    stage_input(sIn, y_buf, (unsigned long long)F_, tid);
    gate_phase(sIn, sWhh, hcur, Wih_d, bias_d,
               c_buf, h_bf + (cur ^ 1) * (B_ * H_), col0, wv, lane);
    cur ^= 1;
    grid_barrier(bar);
  }
}

// ---------------- prep kernels ----------------
__global__ void k_cvt_bf16(const float* __restrict__ src, unsigned short* __restrict__ dst, int n) {
  int i = blockIdx.x * blockDim.x + threadIdx.x;
  if (i < n) dst[i] = f32_to_bf16_rne(src[i]);
}
__global__ void k_bias_add(const float* __restrict__ a, const float* __restrict__ b,
                           float* __restrict__ dst, int n) {
  int i = blockIdx.x * blockDim.x + threadIdx.x;
  if (i < n) dst[i] = a[i] + b[i];
}
__global__ void k_init_state(unsigned short* h, float* c, unsigned* bar) {
  int i = blockIdx.x * blockDim.x + threadIdx.x;
  if (i < 2 * B_ * H_) h[i] = 0;
  if (i < B_ * H_) c[i] = 0.0f;
  if (i == 0) *bar = 0u;
}

// ---------------- launch ----------------
extern "C" void kernel_launch(void* const* d_in, const int* in_sizes, int n_in,
                              void* d_out, int out_size, void* d_ws, size_t ws_size,
                              hipStream_t stream) {
  const float* x     = (const float*)d_in[0];
  const float* Wih_e = (const float*)d_in[1];
  const float* Whh_e = (const float*)d_in[2];
  const float* bih_e = (const float*)d_in[3];
  const float* bhh_e = (const float*)d_in[4];
  const float* Wih_d = (const float*)d_in[5];
  const float* Whh_d = (const float*)d_in[6];
  const float* bih_d = (const float*)d_in[7];
  const float* bhh_d = (const float*)d_in[8];
  const float* Wout  = (const float*)d_in[9];
  const float* bout  = (const float*)d_in[10];
  float* out = (float*)d_out;

  char* ws = (char*)d_ws;
  size_t off = 0;
  auto take = [&](size_t bytes) -> void* {
    void* p = ws + off;
    off += (bytes + 255) & ~(size_t)255;
    return p;
  };
  unsigned*       bar     = (unsigned*)take(256);
  unsigned short* x_bf    = (unsigned short*)take((size_t)B_ * T_ * F_ * 2);
  unsigned short* Wih_e_b = (unsigned short*)take((size_t)G4H_ * F_ * 2);
  unsigned short* Whh_e_b = (unsigned short*)take((size_t)G4H_ * H_ * 2);
  unsigned short* Wih_d_b = (unsigned short*)take((size_t)G4H_ * F_ * 2);
  unsigned short* Whh_d_b = (unsigned short*)take((size_t)G4H_ * H_ * 2);
  unsigned short* Wout_b  = (unsigned short*)take((size_t)F_ * H_ * 2);
  float*          bias_e  = (float*)take((size_t)G4H_ * 4);
  float*          bias_d  = (float*)take((size_t)G4H_ * 4);
  unsigned short* h_bf    = (unsigned short*)take((size_t)2 * B_ * H_ * 2);
  float*          c_buf   = (float*)take((size_t)B_ * H_ * 4);
  unsigned short* y_buf   = (unsigned short*)take((size_t)B_ * F_ * 2);

  auto cvt = [&](const float* s, unsigned short* d, int n) {
    k_cvt_bf16<<<(n + 255) / 256, 256, 0, stream>>>(s, d, n);
  };
  cvt(x,     x_bf,    B_ * T_ * F_);
  cvt(Wih_e, Wih_e_b, G4H_ * F_);
  cvt(Whh_e, Whh_e_b, G4H_ * H_);
  cvt(Wih_d, Wih_d_b, G4H_ * F_);
  cvt(Whh_d, Whh_d_b, G4H_ * H_);
  cvt(Wout,  Wout_b,  F_ * H_);
  k_bias_add<<<(G4H_ + 255) / 256, 256, 0, stream>>>(bih_e, bhh_e, bias_e, G4H_);
  k_bias_add<<<(G4H_ + 255) / 256, 256, 0, stream>>>(bih_d, bhh_d, bias_d, G4H_);
  k_init_state<<<(2 * B_ * H_ + 255) / 256, 256, 0, stream>>>(h_bf, c_buf, bar);

  lstm_ae_persistent<<<NBLK, NTHR, SMEM_ELEMS * 2, stream>>>(
      x_bf, Wih_e_b, Whh_e_b, Wih_d_b, Whh_d_b, Wout_b,
      bias_e, bias_d, bout, h_bf, c_buf, y_buf, out, bar);

  (void)in_sizes; (void)n_in; (void)out_size; (void)ws_size;
}
